// TEncoderlayer_66864050864749
// MI455X (gfx1250) — matrix operand order repaired and checked by reading, not compiled, not run on hardware
//
#include <hip/hip_runtime.h>
#include <hip/hip_bf16.h>

// Sizes from the reference
#define NN    4096
#define DIN   64
#define DOUTC 16
#define CC    4
#define KTOT  (NN * CC)      // 16384  (GEMM K)
#define NOUT  (DOUTC * CC)   // 64     (GEMM N)
#define NWAVE 8              // split-K ways per workgroup
#define KSLC  (KTOT / NWAVE) // 2048 K per wave

typedef __attribute__((ext_vector_type(16))) __bf16       v16bf;
typedef __attribute__((ext_vector_type(8)))  float        v8f;
typedef __attribute__((ext_vector_type(4)))  float        v4f;
typedef __attribute__((ext_vector_type(4)))  unsigned int v4u;

union BUnion { v4u u[2]; v16bf b; };

// ---------------------------------------------------------------------------
// Kernel 1: S[i,r,c] = sum_{j,c1} features[i,j,c1] * weight[j,r,(c-c1)&3]
// then scatter S into the WMMA-B operand, pre-swizzled per 32x16 k/n tile:
//   B[(j,c1)][(r,c')] = S[j,r,(c'-c1)&3]
// Lane layout per tile (contiguous k order, same as kernel 2's A operand):
//   lane = (kin/16)*16 + nin ; element e = kin%16 ; 16 contiguous bf16/lane.
// ---------------------------------------------------------------------------
__global__ __launch_bounds__(256) void build_B_kernel(
    const float* __restrict__ features,
    const float* __restrict__ weight,
    __bf16* __restrict__ Bws)
{
    __shared__ float wsh[DIN * DOUTC * CC];     // 16 KB: whole weight tensor
    for (int t = threadIdx.x; t < DIN * DOUTC * CC; t += 256)
        wsh[t] = weight[t];
    __syncthreads();

    const int idx = blockIdx.x * 256 + threadIdx.x;   // 0 .. 4096*16*4-1
    const int i = idx >> 6;
    const int r = (idx >> 2) & 15;
    const int c = idx & 3;

    const float* frow = features + (size_t)i * (DIN * CC);
    float acc = 0.0f;
#pragma unroll
    for (int c1 = 0; c1 < 4; ++c1) {
        const int wc = (c - c1) & 3;
        float s = 0.0f;
#pragma unroll 8
        for (int j = 0; j < DIN; ++j)
            s += frow[j * 4 + c1] * wsh[j * (DOUTC * CC) + r * 4 + wc];
        acc += s;
    }

    const __bf16 v = (__bf16)acc;
#pragma unroll
    for (int c1 = 0; c1 < 4; ++c1) {
        const int kk = i * 4 + c1;                 // GEMM K index
        const int nn = r * 4 + ((c + c1) & 3);     // GEMM N index
        const int kt  = kk >> 5, kin = kk & 31;
        const int nt  = nn >> 4, nin = nn & 15;
        const int lane = ((kin >> 4) << 4) | nin;
        const int e    = kin & 15;
        Bws[((((kt << 2) | nt) * 32 + lane) << 4) + e] = v;
    }
}

// ---------------------------------------------------------------------------
// Kernel 2: out[i][nn] = sum_k adj_flat[i][k] * B[k][nn] + bias[i][nn]
// M=4096, K=16384, N=64.  One block = one 16-row M tile, 8 waves split K
// (2048 waves total -> ~3+ waves/SIMD to hide NT-stream + B-load latency).
// A streamed once via non-temporal B128 loads; B is 2MB, L2-resident.
// ---------------------------------------------------------------------------
__global__ __launch_bounds__(32 * NWAVE) void gemm_wmma_kernel(
    const float*  __restrict__ adj,
    const __bf16* __restrict__ Bws,
    const float*  __restrict__ bias,
    float*        __restrict__ out)
{
    __shared__ float red[NWAVE * 4 * 8 * 32];   // 32 KB: [wave][ntile][vgpr][lane]

    const int wave  = threadIdx.x >> 5;
    const int lane  = threadIdx.x & 31;
    const int mbase = blockIdx.x * 16;
    const int mrow  = lane & 15;            // A-layout: M = lane%16
    const int h     = lane >> 4;            // A-layout: K half select
    const int kbase = wave * KSLC;          // split-K slice base

    const float* arow = adj + (size_t)(mbase + mrow) * KTOT;

    v8f acc0 = {}, acc1 = {}, acc2 = {}, acc3 = {};

    for (int ks = 0; ks < KSLC; ks += 32) {
        const int kg = kbase + ks;

        // ---- A operand: 16x32 f32 tile, contiguous-K order matching the B packing.
        // lane half h holds K = h*16..h*16+15 (elems 0..15, contiguous)
        const v4f* pa0 = (const v4f*)(arow + kg + h * 16);
        const v4f* pa1 = (const v4f*)(arow + kg + h * 16 + 8);
        v4f a0 = __builtin_nontemporal_load(pa0);
        v4f a1 = __builtin_nontemporal_load(pa0 + 1);
        v4f a2 = __builtin_nontemporal_load(pa1);
        v4f a3 = __builtin_nontemporal_load(pa1 + 1);

        v16bf a;
#pragma unroll
        for (int e = 0; e < 4; ++e) {
            a[e]      = (__bf16)a0[e];
            a[4 + e]  = (__bf16)a1[e];
            a[8 + e]  = (__bf16)a2[e];
            a[12 + e] = (__bf16)a3[e];
        }

        // ---- B operand: pre-swizzled, 2 contiguous B128 loads per lane/tile
        const int kt = kg >> 5;
        const v4u* bbase =
            (const v4u*)(Bws + ((size_t)(kt << 2) * 32 + lane) * 16);
        BUnion b0, b1, b2, b3;
        b0.u[0] = bbase[0];   b0.u[1] = bbase[1];     // nt=0
        b1.u[0] = bbase[64];  b1.u[1] = bbase[65];    // nt=1 (+32 lanes*2)
        b2.u[0] = bbase[128]; b2.u[1] = bbase[129];   // nt=2
        b3.u[0] = bbase[192]; b3.u[1] = bbase[193];   // nt=3

        acc0 = __builtin_amdgcn_wmma_f32_16x16x32_bf16(false, a, false, b0.b,
                                                       (short)0, acc0, false, false);
        acc1 = __builtin_amdgcn_wmma_f32_16x16x32_bf16(false, a, false, b1.b,
                                                       (short)0, acc1, false, false);
        acc2 = __builtin_amdgcn_wmma_f32_16x16x32_bf16(false, a, false, b2.b,
                                                       (short)0, acc2, false, false);
        acc3 = __builtin_amdgcn_wmma_f32_16x16x32_bf16(false, a, false, b3.b,
                                                       (short)0, acc3, false, false);
    }

    // ---- Deterministic split-K reduction through LDS
#pragma unroll
    for (int p = 0; p < 8; ++p) {
        red[((wave * 4 + 0) * 8 + p) * 32 + lane] = acc0[p];
        red[((wave * 4 + 1) * 8 + p) * 32 + lane] = acc1[p];
        red[((wave * 4 + 2) * 8 + p) * 32 + lane] = acc2[p];
        red[((wave * 4 + 3) * 8 + p) * 32 + lane] = acc3[p];
    }
    __syncthreads();

    // C/D layout: VGPR p, lanes 0-15 -> M=p, N=lane ; lanes 16-31 -> M=p+8
    for (int o = threadIdx.x; o < 16 * NOUT; o += 32 * NWAVE) {
        const int row = o >> 6;          // 0..15
        const int col = o & 63;          // 0..63
        const int nt  = col >> 4, nin = col & 15;
        const int p   = row & 7;
        const int l   = ((row >> 3) << 4) | nin;
        float s = 0.0f;
#pragma unroll
        for (int w = 0; w < NWAVE; ++w)
            s += red[((w * 4 + nt) * 8 + p) * 32 + l];
        const size_t oi = (size_t)(mbase + row) * NOUT + col;
        out[oi] = s + bias[oi];
    }
}

extern "C" void kernel_launch(void* const* d_in, const int* in_sizes, int n_in,
                              void* d_out, int out_size, void* d_ws, size_t ws_size,
                              hipStream_t stream) {
    const float* features = (const float*)d_in[0];   // [4096, 64, 4]
    const float* adj      = (const float*)d_in[1];   // [4096, 4096, 4]
    const float* weight   = (const float*)d_in[2];   // [64, 16, 4]
    const float* bias     = (const float*)d_in[3];   // [4096, 16, 4]
    float*       out      = (float*)d_out;           // [4096, 16, 4]
    __bf16*      Bws      = (__bf16*)d_ws;           // 16384 x 64 bf16 = 2 MB

    build_B_kernel<<<(NN * DOUTC * CC) / 256, 256, 0, stream>>>(features, weight, Bws);
    gemm_wmma_kernel<<<NN / 16, 32 * NWAVE, 0, stream>>>(adj, Bws, bias, out);
}
